// EncoderLayer_91010357002686
// MI455X (gfx1250) — compile-verified, matrix operand order checked by reading, not run on hardware
//
#include <hip/hip_runtime.h>
#include <cstdint>

#define DEVFN __device__ __forceinline__

constexpr int S_LEN   = 2048;
constexpr int D_MODEL = 2048;
constexpr int N_HEADS = 16;
constexpr int HEAD_DIM = 128;
constexpr int FFN_DIM = 5632;
constexpr float RMS_EPS = 1e-5f;

// LDS tile: 128 rows x 64 K-elems (bf16), padded row stride 72 u16 (144B).
// 144B keeps every fragment load 16B-aligned (offsets {0,16,32,48}+64*kk) and
// row-start banks 36*r mod 64 cover 16 disjoint 4-bank groups -> conflict-free
// half-wave ds_load_b128.
constexpr int BLK_K      = 64;
constexpr int LDS_STRIDE = 72;
constexpr int TILE_U16   = 128 * LDS_STRIDE;   // 9216 u16 = 18KB per tile

typedef __bf16 v16bf __attribute__((ext_vector_type(16)));
typedef float  v8f   __attribute__((ext_vector_type(8)));
typedef unsigned int u32x4 __attribute__((ext_vector_type(4)));

union FragBF { v16bf v; u32x4 q[2]; };

DEVFN unsigned short f2bf(float f) {
    unsigned int u = __float_as_uint(f);
    u += 0x7fffu + ((u >> 16) & 1u);           // round-to-nearest-even
    return (unsigned short)(u >> 16);
}
DEVFN float bf2f(unsigned short h) { return __uint_as_float(((unsigned int)h) << 16); }

// Load a 16x32 bf16 WMMA fragment from a K-major buffer (row = M/N, K contiguous).
// CDNA5 16-bit A layout: lanes 0-15 hold K {0..7,16..23}, lanes 16-31 hold
// K {8..15,24..31}, row = base_row + (lane & 15).
DEVFN FragBF load_frag(const unsigned short* base, int stride, int row, int kbase, int lane) {
    FragBF f;
    const int r  = row + (lane & 15);
    const int k0 = kbase + ((lane >> 4) << 3);
    const u32x4* p = reinterpret_cast<const u32x4*>(base + (size_t)r * stride + k0);
    f.q[0] = p[0];   // K = k0 .. k0+7
    f.q[1] = p[2];   // K = k0+16 .. k0+23
    return f;
}

DEVFN v8f wmma_bf16(const FragBF& a, const FragBF& b, v8f c) {
    return __builtin_amdgcn_wmma_f32_16x16x32_bf16(
        false, a.v, false, b.v, (short)0, c, false, false);
}

// ---------------------------------------------------------------------------
// Async copy of a 128x64 bf16 tile (global, K-major) into LDS (padded rows),
// using CDNA5 GLOBAL_LOAD_ASYNC_TO_LDS_B128 (tracked on ASYNCcnt, no VGPRs).
// 1024 x 16B chunks; 256 threads issue 4 async b128 transfers each.
// ---------------------------------------------------------------------------
DEVFN void async_copy_tile(const unsigned short* gbase, int stride, int rowbase, int k0,
                           unsigned short* ldsbase, int tid) {
#pragma unroll
    for (int i = 0; i < 4; i++) {
        const int c   = tid + i * 256;         // chunk id 0..1023
        const int row = c >> 3;                // 8 chunks (128B) per row
        const int ko  = (c & 7) * 8;           // u16 offset within row
        const unsigned lds  = (unsigned)(uintptr_t)(ldsbase + row * LDS_STRIDE + ko);
        const unsigned goff = (unsigned)(((size_t)(rowbase + row) * stride + k0 + ko) * 2);
        asm volatile("global_load_async_to_lds_b128 %0, %1, %2 offset:0"
                     :: "v"(lds), "v"(goff), "s"(gbase)
                     : "memory");
    }
}

DEVFN void wait_async_all() {
    asm volatile("s_wait_asynccnt 0" ::: "memory");
}

// ---------------------------------------------------------------------------
// Weight transpose + fp32->bf16 convert: in[rows][cols] fp32 -> out[cols][rows] bf16
// ---------------------------------------------------------------------------
__global__ __launch_bounds__(256) void transpose_cvt_f32_bf16(
    const float* __restrict__ in, unsigned short* __restrict__ out, int rows, int cols) {
    __shared__ float tile[32][33];
    const int n0 = blockIdx.x * 32, k0 = blockIdx.y * 32;
    const int tx = threadIdx.x, ty = threadIdx.y;
#pragma unroll
    for (int j = 0; j < 32; j += 8)
        tile[ty + j][tx] = in[(size_t)(k0 + ty + j) * cols + n0 + tx];
    __syncthreads();
#pragma unroll
    for (int j = 0; j < 32; j += 8)
        out[(size_t)(n0 + ty + j) * rows + k0 + tx] = f2bf(tile[tx][ty + j]);
}

// Per-head V transpose (bf16): v[S][D] -> vt[H][HD][S]
__global__ __launch_bounds__(256) void transpose_v_kernel(
    const unsigned short* __restrict__ v, unsigned short* __restrict__ vt) {
    __shared__ unsigned short tile[32][33];
    const int s0 = blockIdx.x * 32, d0 = blockIdx.y * 32, head = blockIdx.z;
    const int tx = threadIdx.x, ty = threadIdx.y;
#pragma unroll
    for (int j = 0; j < 32; j += 8)
        tile[ty + j][tx] = v[(size_t)(s0 + ty + j) * D_MODEL + head * HEAD_DIM + d0 + tx];
    __syncthreads();
#pragma unroll
    for (int j = 0; j < 32; j += 8)
        vt[(size_t)head * HEAD_DIM * S_LEN + (size_t)(d0 + ty + j) * S_LEN + s0 + tx] =
            tile[tx][ty + j];
}

// ---------------------------------------------------------------------------
// RMSNorm: fp32 row -> bf16 normalized row
// ---------------------------------------------------------------------------
__global__ __launch_bounds__(256) void rmsnorm_kernel(
    const float* __restrict__ x, const float* __restrict__ w,
    unsigned short* __restrict__ out) {
    const int row = blockIdx.x;
    const float* xr = x + (size_t)row * D_MODEL;
    float s = 0.0f;
    for (int i = threadIdx.x; i < D_MODEL; i += 256) { float v = xr[i]; s += v * v; }
    __shared__ float red[256];
    red[threadIdx.x] = s;
    __syncthreads();
    for (int off = 128; off > 0; off >>= 1) {
        if (threadIdx.x < off) red[threadIdx.x] += red[threadIdx.x + off];
        __syncthreads();
    }
    const float inv = rsqrtf(red[0] / (float)D_MODEL + RMS_EPS);
    for (int i = threadIdx.x; i < D_MODEL; i += 256)
        out[(size_t)row * D_MODEL + i] = f2bf(xr[i] * inv * w[i]);
}

// ---------------------------------------------------------------------------
// RoPE on bf16 [S, H, HD]; scale folds 1/sqrt(HD) into q.
// ---------------------------------------------------------------------------
__global__ __launch_bounds__(256) void rope_kernel(
    const unsigned short* __restrict__ in, unsigned short* __restrict__ out,
    const float* __restrict__ cosb, const float* __restrict__ sinb, float scale) {
    const int idx = blockIdx.x * 256 + threadIdx.x;     // S*D/2 threads
    const int s   = idx / (D_MODEL / 2);
    const int rem = idx % (D_MODEL / 2);
    const int h   = rem / (HEAD_DIM / 2);
    const int i   = rem % (HEAD_DIM / 2);
    const size_t base = (size_t)s * D_MODEL + h * HEAD_DIM + 2 * i;
    const float re = bf2f(in[base]), im = bf2f(in[base + 1]);
    const float c  = cosb[(size_t)s * (HEAD_DIM / 2) + i];
    const float sn = sinb[(size_t)s * (HEAD_DIM / 2) + i];
    out[base]     = f2bf((re * c - im * sn) * scale);
    out[base + 1] = f2bf((re * sn + im * c) * scale);
}

// ---------------------------------------------------------------------------
// bf16 WMMA GEMM with async-LDS double buffering, BLK_K=64 per stage.
// C[M,N] = A[M,K] (K-major) * B[N,K] (K-major). Block 128x128, 8 waves,
// wave tile 64x32 (4x2 wmma tiles), 2 kk sub-steps per stage.
// EPI: 0 = bf16 out, 1 = fp32 out + fp32 residual, 2 = SwiGLU(acc0, acc1)->bf16
// ---------------------------------------------------------------------------
template <int EPI, bool DUAL>
__global__ __launch_bounds__(256) void gemm_bf16_kernel(
    const unsigned short* __restrict__ A,
    const unsigned short* __restrict__ B0,
    const unsigned short* __restrict__ B1,
    unsigned short* __restrict__ outb,
    float* __restrict__ outf,
    const float* __restrict__ resid,
    int M, int N, int K) {
    extern __shared__ unsigned short smem[];
    unsigned short* sA  = smem;                   // [2][128][72]
    unsigned short* sB0 = smem + 2 * TILE_U16;    // [2][128][72]
    unsigned short* sB1 = smem + 4 * TILE_U16;    // [2][128][72] (DUAL only)

    const int tid   = threadIdx.x;
    const int lane  = tid & 31;
    const int w     = tid >> 5;
    const int mloc  = (w >> 2) * 64;              // wave M offset within block tile
    const int nloc  = (w & 3) * 32;               // wave N offset within block tile
    const int mblk  = blockIdx.y * 128;
    const int nblk  = blockIdx.x * 128;
    (void)M;

    v8f acc0[4][2];
    v8f acc1[4][2];
#pragma unroll
    for (int mi = 0; mi < 4; mi++)
#pragma unroll
        for (int ni = 0; ni < 2; ni++) { acc0[mi][ni] = {}; acc1[mi][ni] = {}; }

    // prefetch first k-tile into buffer 0
    async_copy_tile(A,  K, mblk, 0, sA,  tid);
    async_copy_tile(B0, K, nblk, 0, sB0, tid);
    if constexpr (DUAL) async_copy_tile(B1, K, nblk, 0, sB1, tid);

    for (int k0 = 0; k0 < K; k0 += BLK_K) {
        const int buf = (k0 >> 6) & 1;
        wait_async_all();
        __syncthreads();
        if (k0 + BLK_K < K) {
            const int nb = buf ^ 1;
            async_copy_tile(A,  K, mblk, k0 + BLK_K, sA  + nb * TILE_U16, tid);
            async_copy_tile(B0, K, nblk, k0 + BLK_K, sB0 + nb * TILE_U16, tid);
            if constexpr (DUAL)
                async_copy_tile(B1, K, nblk, k0 + BLK_K, sB1 + nb * TILE_U16, tid);
        }
        const unsigned short* tA  = sA  + buf * TILE_U16;
        const unsigned short* tB0 = sB0 + buf * TILE_U16;
        const unsigned short* tB1 = sB1 + buf * TILE_U16;

#pragma unroll
        for (int kk = 0; kk < BLK_K; kk += 32) {
            FragBF a[4], b[2], c[2];
#pragma unroll
            for (int mi = 0; mi < 4; mi++)
                a[mi] = load_frag(tA, LDS_STRIDE, mloc + mi * 16, kk, lane);
#pragma unroll
            for (int ni = 0; ni < 2; ni++)
                b[ni] = load_frag(tB0, LDS_STRIDE, nloc + ni * 16, kk, lane);
            if constexpr (DUAL) {
#pragma unroll
                for (int ni = 0; ni < 2; ni++)
                    c[ni] = load_frag(tB1, LDS_STRIDE, nloc + ni * 16, kk, lane);
            }
#pragma unroll
            for (int mi = 0; mi < 4; mi++)
#pragma unroll
                for (int ni = 0; ni < 2; ni++) {
                    acc0[mi][ni] = wmma_bf16(a[mi], b[ni], acc0[mi][ni]);
                    if constexpr (DUAL) acc1[mi][ni] = wmma_bf16(a[mi], c[ni], acc1[mi][ni]);
                }
        }
    }

    const int rowadd = (lane & 16) ? 8 : 0;
#pragma unroll
    for (int mi = 0; mi < 4; mi++)
#pragma unroll
        for (int ni = 0; ni < 2; ni++)
#pragma unroll
            for (int r = 0; r < 8; r++) {
                const int row = mblk + mloc + mi * 16 + r + rowadd;
                const int col = nblk + nloc + ni * 16 + (lane & 15);
                const size_t idx = (size_t)row * N + col;
                const float v0 = acc0[mi][ni][r];
                if constexpr (EPI == 0) {
                    outb[idx] = f2bf(v0);
                } else if constexpr (EPI == 1) {
                    outf[idx] = resid[idx] + v0;
                } else {
                    const float g = v0 / (1.0f + __expf(-v0));   // silu
                    outb[idx] = f2bf(g * acc1[mi][ni][r]);
                }
            }
}

// ---------------------------------------------------------------------------
// Flash attention (causal). Grid (S/128, H). 8 waves: wave w owns 16 q-rows.
// Q,K bf16 row-major [S,D]; Vt bf16 [H][HD][S]. Scale pre-folded into Q.
// ---------------------------------------------------------------------------
__global__ __launch_bounds__(256) void flash_attn_kernel(
    const unsigned short* __restrict__ Q,
    const unsigned short* __restrict__ Kb,
    const unsigned short* __restrict__ Vt,
    unsigned short* __restrict__ O) {
    __shared__ __align__(16) unsigned short ldsP[8][16][128];
    const int lane = threadIdx.x & 31;
    const int w    = threadIdx.x >> 5;
    const int qb   = blockIdx.x * 128;
    const int head = blockIdx.y;
    const int rowadd = (lane & 16) ? 8 : 0;

    FragBF qf[4];
#pragma unroll
    for (int kk = 0; kk < 4; kk++)
        qf[kk] = load_frag(Q, D_MODEL, qb + w * 16, head * HEAD_DIM + kk * 32, lane);

    v8f oacc[8];
#pragma unroll
    for (int nt = 0; nt < 8; nt++) oacc[nt] = {};
    float m_s[8], l_s[8];
    int qrow[8];
#pragma unroll
    for (int r = 0; r < 8; r++) {
        m_s[r] = -3.0e38f; l_s[r] = 0.0f;
        qrow[r] = qb + w * 16 + r + rowadd;
    }

    const int nkt = blockIdx.x + 1;   // causal: only key tiles <= q tile
    for (int kt = 0; kt < nkt; kt++) {
        v8f sacc[8];
#pragma unroll
        for (int nt = 0; nt < 8; nt++) sacc[nt] = {};
#pragma unroll
        for (int nt = 0; nt < 8; nt++)
#pragma unroll
            for (int kk = 0; kk < 4; kk++) {
                FragBF kf = load_frag(Kb, D_MODEL, kt * 128 + nt * 16,
                                      head * HEAD_DIM + kk * 32, lane);
                sacc[nt] = wmma_bf16(qf[kk], kf, sacc[nt]);
            }

        // causal mask + per-row running max
        float pm[8];
#pragma unroll
        for (int r = 0; r < 8; r++) pm[r] = -3.0e38f;
#pragma unroll
        for (int nt = 0; nt < 8; nt++) {
            const int key = kt * 128 + nt * 16 + (lane & 15);
#pragma unroll
            for (int r = 0; r < 8; r++) {
                float v = sacc[nt][r];
                if (key > qrow[r]) v = -3.0e38f;
                sacc[nt][r] = v;
                pm[r] = fmaxf(pm[r], v);
            }
        }
#pragma unroll
        for (int r = 0; r < 8; r++) {
            float v = pm[r];
            v = fmaxf(v, __shfl_xor(v, 1, 16));
            v = fmaxf(v, __shfl_xor(v, 2, 16));
            v = fmaxf(v, __shfl_xor(v, 4, 16));
            v = fmaxf(v, __shfl_xor(v, 8, 16));
            pm[r] = v;
        }
        float corr[8];
#pragma unroll
        for (int r = 0; r < 8; r++) {
            const float mn = fmaxf(m_s[r], pm[r]);
            corr[r] = __expf(m_s[r] - mn);
            m_s[r]  = mn;
        }
        float rsum[8];
#pragma unroll
        for (int r = 0; r < 8; r++) rsum[r] = 0.0f;
#pragma unroll
        for (int nt = 0; nt < 8; nt++)
#pragma unroll
            for (int r = 0; r < 8; r++) {
                const float p = __expf(sacc[nt][r] - m_s[r]);
                sacc[nt][r] = p;
                rsum[r] += p;
            }
#pragma unroll
        for (int r = 0; r < 8; r++) {
            float v = rsum[r];
            v += __shfl_xor(v, 1, 16);
            v += __shfl_xor(v, 2, 16);
            v += __shfl_xor(v, 4, 16);
            v += __shfl_xor(v, 8, 16);
            l_s[r] = l_s[r] * corr[r] + v;
        }
#pragma unroll
        for (int nt = 0; nt < 8; nt++)
#pragma unroll
            for (int r = 0; r < 8; r++) oacc[nt][r] *= corr[r];

        // stage P into LDS (accumulator layout -> row-major bf16 16x128 per wave)
        __syncthreads();
#pragma unroll
        for (int nt = 0; nt < 8; nt++)
#pragma unroll
            for (int r = 0; r < 8; r++)
                ldsP[w][r + rowadd][nt * 16 + (lane & 15)] = f2bf(sacc[nt][r]);
        __syncthreads();

        // O += P * V  (A = P from LDS, B = Vt rows, K = key index)
        const unsigned short* pbase = &ldsP[w][0][0];
        const unsigned short* vbase = Vt + (size_t)head * HEAD_DIM * S_LEN;
        FragBF pa[4];
#pragma unroll
        for (int kk = 0; kk < 4; kk++)
            pa[kk] = load_frag(pbase, 128, 0, kk * 32, lane);
#pragma unroll
        for (int nto = 0; nto < 8; nto++)
#pragma unroll
            for (int kk = 0; kk < 4; kk++) {
                FragBF vb = load_frag(vbase, S_LEN, nto * 16, kt * 128 + kk * 32, lane);
                oacc[nto] = wmma_bf16(pa[kk], vb, oacc[nto]);
            }
    }

#pragma unroll
    for (int r = 0; r < 8; r++) l_s[r] = 1.0f / l_s[r];
#pragma unroll
    for (int nt = 0; nt < 8; nt++)
#pragma unroll
        for (int r = 0; r < 8; r++)
            O[(size_t)qrow[r] * D_MODEL + head * HEAD_DIM + nt * 16 + (lane & 15)] =
                f2bf(oacc[nt][r] * l_s[r]);
}

// ---------------------------------------------------------------------------
// Launcher
// ---------------------------------------------------------------------------
extern "C" void kernel_launch(void* const* d_in, const int* in_sizes, int n_in,
                              void* d_out, int out_size, void* d_ws, size_t ws_size,
                              hipStream_t stream) {
    (void)in_sizes; (void)n_in; (void)out_size; (void)ws_size;
    const float* x      = (const float*)d_in[0];
    const float* fcos   = (const float*)d_in[1];
    const float* fsin   = (const float*)d_in[2];
    /* d_in[3] = mask (causal hardcoded) */
    const float* wq     = (const float*)d_in[4];
    const float* wk     = (const float*)d_in[5];
    const float* wv     = (const float*)d_in[6];
    const float* wo     = (const float*)d_in[7];
    const float* w1     = (const float*)d_in[8];
    const float* w2     = (const float*)d_in[9];
    const float* w3     = (const float*)d_in[10];
    const float* attn_w = (const float*)d_in[11];
    const float* ffn_w  = (const float*)d_in[12];
    float* out = (float*)d_out;

    char* ws = (char*)d_ws;
    size_t off = 0;
    auto alloc_u16 = [&](size_t n) { unsigned short* p = (unsigned short*)(ws + off); off += n * 2; return p; };
    auto alloc_f32 = [&](size_t n) { float* p = (float*)(ws + off); off += n * 4; return p; };

    const size_t DD = (size_t)D_MODEL * D_MODEL;
    const size_t SD = (size_t)S_LEN * D_MODEL;
    const size_t DF = (size_t)D_MODEL * FFN_DIM;
    const size_t SF = (size_t)S_LEN * FFN_DIM;

    unsigned short* wqt  = alloc_u16(DD);
    unsigned short* wkt  = alloc_u16(DD);
    unsigned short* wvt  = alloc_u16(DD);
    unsigned short* wot  = alloc_u16(DD);
    unsigned short* w1t  = alloc_u16(DF);
    unsigned short* w3t  = alloc_u16(DF);
    unsigned short* w2t  = alloc_u16(DF);
    unsigned short* xn   = alloc_u16(SD);
    unsigned short* qbuf = alloc_u16(SD);
    unsigned short* kbuf = alloc_u16(SD);
    unsigned short* vbuf = alloc_u16(SD);
    unsigned short* qr   = alloc_u16(SD);
    unsigned short* kr   = alloc_u16(SD);
    unsigned short* vt   = alloc_u16(SD);
    unsigned short* attn = alloc_u16(SD);
    unsigned short* hn   = alloc_u16(SD);
    unsigned short* gate = alloc_u16(SF);
    float* hbuf = alloc_f32(SD);

    const size_t lds_single = 4 * TILE_U16 * sizeof(unsigned short);  // 72KB
    const size_t lds_dual   = 6 * TILE_U16 * sizeof(unsigned short);  // 108KB

    const dim3 tb(32, 8);
    // weights -> bf16, K-major
    transpose_cvt_f32_bf16<<<dim3(64, 64), tb, 0, stream>>>(wq, wqt, D_MODEL, D_MODEL);
    transpose_cvt_f32_bf16<<<dim3(64, 64), tb, 0, stream>>>(wk, wkt, D_MODEL, D_MODEL);
    transpose_cvt_f32_bf16<<<dim3(64, 64), tb, 0, stream>>>(wv, wvt, D_MODEL, D_MODEL);
    transpose_cvt_f32_bf16<<<dim3(64, 64), tb, 0, stream>>>(wo, wot, D_MODEL, D_MODEL);
    transpose_cvt_f32_bf16<<<dim3(176, 64), tb, 0, stream>>>(w1, w1t, D_MODEL, FFN_DIM);
    transpose_cvt_f32_bf16<<<dim3(176, 64), tb, 0, stream>>>(w3, w3t, D_MODEL, FFN_DIM);
    transpose_cvt_f32_bf16<<<dim3(64, 176), tb, 0, stream>>>(w2, w2t, FFN_DIM, D_MODEL);

    // attention branch
    rmsnorm_kernel<<<S_LEN, 256, 0, stream>>>(x, attn_w, xn);
    gemm_bf16_kernel<0, false><<<dim3(16, 16), 256, lds_single, stream>>>(
        xn, wqt, nullptr, qbuf, nullptr, nullptr, S_LEN, D_MODEL, D_MODEL);
    gemm_bf16_kernel<0, false><<<dim3(16, 16), 256, lds_single, stream>>>(
        xn, wkt, nullptr, kbuf, nullptr, nullptr, S_LEN, D_MODEL, D_MODEL);
    gemm_bf16_kernel<0, false><<<dim3(16, 16), 256, lds_single, stream>>>(
        xn, wvt, nullptr, vbuf, nullptr, nullptr, S_LEN, D_MODEL, D_MODEL);

    rope_kernel<<<8192, 256, 0, stream>>>(qbuf, qr, fcos, fsin, 0.08838834764831845f); // 1/sqrt(128)
    rope_kernel<<<8192, 256, 0, stream>>>(kbuf, kr, fcos, fsin, 1.0f);
    transpose_v_kernel<<<dim3(64, 4, 16), tb, 0, stream>>>(vbuf, vt);

    flash_attn_kernel<<<dim3(16, 16), 256, 0, stream>>>(qr, kr, vt, attn);

    gemm_bf16_kernel<1, false><<<dim3(16, 16), 256, lds_single, stream>>>(
        attn, wot, nullptr, nullptr, hbuf, x, S_LEN, D_MODEL, D_MODEL);

    // FFN branch
    rmsnorm_kernel<<<S_LEN, 256, 0, stream>>>(hbuf, ffn_w, hn);
    gemm_bf16_kernel<2, true><<<dim3(44, 16), 256, lds_dual, stream>>>(
        hn, w1t, w3t, gate, nullptr, nullptr, S_LEN, FFN_DIM, D_MODEL);
    gemm_bf16_kernel<1, false><<<dim3(16, 16), 256, lds_single, stream>>>(
        gate, w2t, nullptr, nullptr, out, hbuf, S_LEN, D_MODEL, FFN_DIM);
}
